// SinNO2d_22539988369880
// MI455X (gfx1250) — matrix operand order repaired and checked by reading, not compiled, not run on hardware
//
#include <hip/hip_runtime.h>
#if __has_include(<hip/amd_detail/amd_gfx1250_TDM.h>)
#define TDM_6ARG 1
#endif

typedef __attribute__((ext_vector_type(16))) _Float16 v16h;
typedef __attribute__((ext_vector_type(8)))  _Float16 v8h;
typedef __attribute__((ext_vector_type(8)))  float    v8f;
typedef __attribute__((ext_vector_type(4)))  float    vf4;
typedef __attribute__((ext_vector_type(4)))  unsigned v4u;
typedef __attribute__((ext_vector_type(8)))  int      v8i;
typedef __attribute__((ext_vector_type(4)))  int      v4i;

// ---- workspace layout (bytes) ----
#define WS_SFWD 0                          // _Float16[128*128]       fwd DST matrix
#define WS_SINV (128*128*2)                // _Float16[128*128]       inv DST matrix (scaled)
#define WS_A    (2*128*128*2)              // _Float16[512][80][80]   spectrum, [img][col v][row u]
#define WS_BM   (WS_A + 512*80*80*2)       // _Float16[64*64][32][16] bm, [m1*64+m2][o][b]

__device__ __forceinline__ v8f wmma16x16x32(v16h a, v16h b, v8f c) {
  return __builtin_amdgcn_wmma_f32_16x16x32_f16(false, a, false, b, (short)0, c,
                                                false, false);
}
__device__ __forceinline__ v16h cat8(v8h lo, v8h hi) {
  return __builtin_shufflevector(lo, hi, 0,1,2,3,4,5,6,7,8,9,10,11,12,13,14,15);
}
__device__ __forceinline__ v8h packh(v8f a) {
  v8h h;
#pragma unroll
  for (int r = 0; r < 8; ++r) h[r] = (_Float16)a[r];
  return h;
}

// -------------------- DST-I matrices --------------------
__global__ void build_s_kernel(_Float16* __restrict__ sf, _Float16* __restrict__ si) {
  int idx = blockIdx.x * 256 + threadIdx.x;   // 0..16383
  int r = idx >> 7, c = idx & 127;
  float v = sinf(3.14159265358979323846f * (float)((r + 1) * (c + 1)) / 129.0f);
  sf[idx] = (_Float16)v;
  si[idx] = (_Float16)(v * (2.0f / 129.0f));
}

// -------------------- forward DST, truncated, stored transposed [v][u] ------------
__global__ __launch_bounds__(256)
void dst_fwd_kernel(const float* __restrict__ u,
                    const _Float16* __restrict__ sf,
                    _Float16* __restrict__ aws) {
  __shared__ _Float16 u_s[128 * 128];   // 32 KB  input image (f16), [x][y]
  __shared__ _Float16 t_s[80 * 128];    // 20 KB  stage-1 result, [v][x]
  const int img = blockIdx.x;           // b*32 + c
  const float* up = u + (size_t)img * (128 * 128);
  for (int i = threadIdx.x; i < 4096; i += 256) {
    vf4 f = ((const vf4*)up)[i];
    u_s[4*i+0] = (_Float16)f[0]; u_s[4*i+1] = (_Float16)f[1];
    u_s[4*i+2] = (_Float16)f[2]; u_s[4*i+3] = (_Float16)f[3];
  }
  __syncthreads();

  const int wave = threadIdx.x >> 5;
  const int lane = threadIdx.x & 31;
  const int mrow = lane & 15;           // A row / B col / D col
  const int klo  = (lane & 16) >> 1;    // D row offset
  const int kb0  = lane & 16;           // B K offset

  // stage 1: T[x,v] = sum_y U[x,y]*S[y,v]  (M=x 8 tiles, N=v 5 tiles, K=128)
  const int x0 = wave * 16;
  for (int vt = 0; vt < 5; ++vt) {
    const int v0 = vt * 16;
    v8f acc = {};
    for (int kc = 0; kc < 4; ++kc) {
      const int y0 = kc * 32;
      v8h alo = *(const v8h*)&u_s[(x0 + mrow) * 128 + y0 + kb0];
      v8h ahi = *(const v8h*)&u_s[(x0 + mrow) * 128 + y0 + kb0 + 8];
      // B[K=y][N=v] = S[y][v] = S[v][y]  -> K-contiguous
      v16h b = *(const v16h*)&sf[(v0 + mrow) * 128 + y0 + kb0];
      acc = wmma16x16x32(cat8(alo, ahi), b, acc);
    }
    *(v8h*)&t_s[(v0 + mrow) * 128 + x0 + klo] = packh(acc);   // [v][x]
  }
  __syncthreads();

  // stage 2: A[u,v] = sum_x S[x,u]*T[x,v]  (M=u 5 tiles, N=v 5 tiles, K=128)
  _Float16* aout = aws + (size_t)img * (80 * 80);
  for (int t = wave; t < 25; t += 8) {
    const int u0 = (t / 5) * 16;
    const int v0 = (t % 5) * 16;
    v8f acc = {};
    for (int kc = 0; kc < 4; ++kc) {
      const int xk = kc * 32;
      // A[M=u][K=x] = S[x][u] = S[u][x] -> K-contiguous
      v8h alo = *(const v8h*)&sf[(u0 + mrow) * 128 + xk + kb0];
      v8h ahi = *(const v8h*)&sf[(u0 + mrow) * 128 + xk + kb0 + 8];
      v16h b = *(const v16h*)&t_s[(v0 + mrow) * 128 + xk + kb0];
      acc = wmma16x16x32(cat8(alo, ahi), b, acc);
    }
    *(v8h*)&aout[(v0 + mrow) * 80 + u0 + klo] = packh(acc);   // [v][u]
  }
}

// -------------------- per-mode contraction (weight-stream bound) --------------------
// bm[m1,m2][o][b] = sum_i a[b, c(i), m1+ki(i), m2+kj(i)] * w[i,o,m1*64+m2], K=800=25*32
__global__ __launch_bounds__(256)
void mode_gemm_kernel(const float* __restrict__ w,
                      const _Float16* __restrict__ aws,
                      _Float16* __restrict__ bm16) {
  __shared__ _Float16 w_t[16 * 32 * 32];     // 32 KB [mode][o][k]  (B fragment order)
  __shared__ _Float16 a_frag[16 * 32 * 16];  // 16 KB [mode][lane][e] (A fragment order)
  const int m1    = blockIdx.x >> 2;
  const int m2_0  = (blockIdx.x & 3) * 16;
  const int xbase = m1 * 64 + m2_0;
  const int wave  = threadIdx.x >> 5;
  const int lane  = threadIdx.x & 31;
  const int mrow  = lane & 15;
  const int klo   = (lane & 16) >> 1;
  const int kb0   = lane & 16;

  v8f acc00 = {}, acc01 = {}, acc10 = {}, acc11 = {};

  for (int t = 0; t < 25; ++t) {
    const int kbase = t * 32;
    __syncthreads();
    // stage 32 KB weight slab: coalesced float4, single-use -> non-temporal
    for (int idx = threadIdx.x; idx < 4096; idx += 256) {
      const int g = idx & 3;
      const int pair = idx >> 2;
      const int k = pair >> 5, o = pair & 31;
      const int i = kbase + k;
      const vf4 f4 = __builtin_nontemporal_load(
          (const vf4*)(w + (size_t)(i * 32 + o) * 4096 + xbase) + g);
      const int mb = g * 4;
      w_t[((mb + 0) * 32 + o) * 32 + k] = (_Float16)f4[0];
      w_t[((mb + 1) * 32 + o) * 32 + k] = (_Float16)f4[1];
      w_t[((mb + 2) * 32 + o) * 32 + k] = (_Float16)f4[2];
      w_t[((mb + 3) * 32 + o) * 32 + k] = (_Float16)f4[3];
    }
    // gather A fragments (unfold) straight into WMMA lane order
    for (int idx = threadIdx.x; idx < 8192; idx += 256) {
      const int e  = idx & 15;
      const int ln = (idx >> 4) & 31;
      const int mm = idx >> 9;
      const int K  = (ln & 16) | e;
      const int i  = kbase + K;
      const int c  = i / 25;
      const int rr = i - c * 25;
      const int ki = rr / 5;
      const int kj = rr - ki * 5;
      const int b  = ln & 15;
      a_frag[idx] =
          aws[(size_t)(b * 32 + c) * 6400 + (m2_0 + mm + kj) * 80 + (m1 + ki)];
    }
    __syncthreads();
#pragma unroll
    for (int mi = 0; mi < 2; ++mi) {
      const int mm = wave * 2 + mi;
      const v16h a  = *(const v16h*)&a_frag[(mm * 32 + lane) * 16];
      const v16h b0 = *(const v16h*)&w_t[(mm * 32 + mrow) * 32 + kb0];
      const v16h b1 = *(const v16h*)&w_t[(mm * 32 + 16 + mrow) * 32 + kb0];
      if (mi == 0) { acc00 = wmma16x16x32(a, b0, acc00);
                     acc01 = wmma16x16x32(a, b1, acc01); }
      else         { acc10 = wmma16x16x32(a, b0, acc10);
                     acc11 = wmma16x16x32(a, b1, acc11); }
    }
  }
  // store bm16[(m1*64+m2)*512 + o*16 + b] -- b contiguous -> packed b128 stores
#pragma unroll
  for (int mi = 0; mi < 2; ++mi) {
    const int m2 = m2_0 + wave * 2 + mi;
#pragma unroll
    for (int nt = 0; nt < 2; ++nt) {
      const int o = nt * 16 + mrow;
      const v8f acc = mi == 0 ? (nt == 0 ? acc00 : acc01)
                              : (nt == 0 ? acc10 : acc11);
      *(v8h*)&bm16[(size_t)(m1 * 64 + m2) * 512 + o * 16 + klo] = packh(acc);
    }
  }
}

// -------------------- inverse DST (64x64 nonzero -> 128x128) --------------------
__global__ __launch_bounds__(256)
void dst_inv_kernel(const _Float16* __restrict__ bm16,
                    const _Float16* __restrict__ si,
                    float* __restrict__ out) {
  __shared__ _Float16 si_s[128 * 128];   // 32 KB, loaded by Tensor Data Mover
  __shared__ _Float16 bm_s[64 * 64];     //  8 KB, [u][v]
  __shared__ _Float16 t_s[128 * 64];     // 16 KB, [y][u]
  const int img  = blockIdx.x;           // b*32 + o
  const int bidx = img >> 5, oidx = img & 31;

  if ((threadIdx.x >> 5) == 0) {
    // D#: 2-D tensor, 128x128 tile, data_size=2B, type=2 ("image")
    unsigned long long ga = (unsigned long long)(const void*)si;
    unsigned lds = (unsigned)(unsigned long long)(void*)si_s;
    v4u g0; g0[0] = 1u;                                  // count=1
    g0[1] = lds;                                         // lds_addr
    g0[2] = (unsigned)(ga & 0xffffffffu);                // global_addr lo
    g0[3] = (unsigned)((ga >> 32) & 0x1ffffffu) | (2u << 30); // ga hi | type=2
    v8i g1;
    g1[0] = 0x00010000;          // workgroup_mask=0, data_size=1 (2 bytes)
    g1[1] = (int)(128u << 16);   // tensor_dim0 = 128 (bits 63:48)
    g1[2] = (int)(128u << 16);   // tensor_dim1 = 128 (bits 95:80)
    g1[3] = (int)(128u << 16);   // tile_dim0   = 128 (bits 127:112)
    g1[4] = 128;                 // tile_dim1 = 128, tile_dim2 = 0
    g1[5] = 128;                 // tensor_dim0_stride = 128
    g1[6] = 0; g1[7] = 0;
    v4i gz; gz[0] = 0; gz[1] = 0; gz[2] = 0; gz[3] = 0;
#ifdef TDM_6ARG
    v8i gz8; gz8[0]=0; gz8[1]=0; gz8[2]=0; gz8[3]=0; gz8[4]=0; gz8[5]=0; gz8[6]=0; gz8[7]=0;
    __builtin_amdgcn_tensor_load_to_lds(g0, g1, gz, gz, gz8, 0);
#else
    __builtin_amdgcn_tensor_load_to_lds(g0, g1, gz, gz, 0);
#endif
    __builtin_amdgcn_s_wait_tensorcnt(0);
  }
  for (int idx = threadIdx.x; idx < 4096; idx += 256)
    bm_s[idx] = bm16[(size_t)idx * 512 + oidx * 16 + bidx];
  __syncthreads();

  const int wave = threadIdx.x >> 5;
  const int lane = threadIdx.x & 31;
  const int mrow = lane & 15;
  const int klo  = (lane & 16) >> 1;
  const int kb0  = lane & 16;

  // stage 1: T[u,y] = sum_v bm[u,v]*Si[v,y]  (M=u 4 tiles, N=y 8 tiles, K=64)
  for (int j = 0; j < 4; ++j) {
    const int tt = wave * 4 + j;
    const int u0 = (tt >> 3) * 16;
    const int y0 = (tt & 7) * 16;
    v8f acc = {};
    for (int kc = 0; kc < 2; ++kc) {
      const int v0 = kc * 32;
      v8h alo = *(const v8h*)&bm_s[(u0 + mrow) * 64 + v0 + kb0];
      v8h ahi = *(const v8h*)&bm_s[(u0 + mrow) * 64 + v0 + kb0 + 8];
      v16h b = *(const v16h*)&si_s[(y0 + mrow) * 128 + v0 + kb0]; // Si[v][y]=Si[y][v]
      acc = wmma16x16x32(cat8(alo, ahi), b, acc);
    }
    *(v8h*)&t_s[(y0 + mrow) * 64 + u0 + klo] = packh(acc);        // [y][u]
  }
  __syncthreads();

  // stage 2 (transposed tile: M=y, N=x): out[x,y] = sum_u Si[u,x]*T[u,y]
  float* op = out + (size_t)img * (128 * 128);
  for (int j = 0; j < 8; ++j) {
    const int y0 = wave * 16;
    const int x0 = j * 16;
    v8f acc = {};
    for (int kc = 0; kc < 2; ++kc) {
      const int u0 = kc * 32;
      v8h alo = *(const v8h*)&t_s[(y0 + mrow) * 64 + u0 + kb0];   // A[y][u]=T[u][y]
      v8h ahi = *(const v8h*)&t_s[(y0 + mrow) * 64 + u0 + kb0 + 8];
      v16h b = *(const v16h*)&si_s[(x0 + mrow) * 128 + u0 + kb0]; // Si[u][x]=Si[x][u]
      acc = wmma16x16x32(cat8(alo, ahi), b, acc);
    }
    // D[M=y][N=x] -> out[(x0+mrow)*128 + y0+klo+r]: contiguous -> 2x float4
    vf4 lo, hi;
#pragma unroll
    for (int r = 0; r < 4; ++r) { lo[r] = acc[r]; hi[r] = acc[r + 4]; }
    *(vf4*)&op[(x0 + mrow) * 128 + y0 + klo]     = lo;
    *(vf4*)&op[(x0 + mrow) * 128 + y0 + klo + 4] = hi;
  }
}

extern "C" void kernel_launch(void* const* d_in, const int* in_sizes, int n_in,
                              void* d_out, int out_size, void* d_ws, size_t ws_size,
                              hipStream_t stream) {
  const float* u = (const float*)d_in[0];        // (16,32,128,128) f32
  const float* w = (const float*)d_in[1];        // (800,32,4096)   f32
  float* out = (float*)d_out;                    // (16,32,128,128) f32
  char* ws = (char*)d_ws;
  _Float16* sf   = (_Float16*)(ws + WS_SFWD);
  _Float16* si   = (_Float16*)(ws + WS_SINV);
  _Float16* aws  = (_Float16*)(ws + WS_A);
  _Float16* bm16 = (_Float16*)(ws + WS_BM);

  build_s_kernel<<<64, 256, 0, stream>>>(sf, si);
  dst_fwd_kernel<<<512, 256, 0, stream>>>(u, sf, aws);
  mode_gemm_kernel<<<256, 256, 0, stream>>>(w, aws, bm16);
  dst_inv_kernel<<<512, 256, 0, stream>>>(bm16, si, out);
}
